// STHGCN_84464826843684
// MI455X (gfx1250) — compile-verified, matrix operand order checked by reading, not run on hardware
//
#include <hip/hip_runtime.h>
#include <hip/hip_bf16.h>

#define D 128
#define POI 8192
#define N_HE 51
#define N_UC 42
#define N_UI 8234   // POI + N_UC
#define KP   8240   // padded f16 row stride for U_I (multiple of 8 halves = 16B)

typedef __attribute__((ext_vector_type(16))) _Float16 v16h;
typedef __attribute__((ext_vector_type(8)))  _Float16 v8h;
typedef __attribute__((ext_vector_type(8)))  float    v8f;

// 16-bit 16x32 WMMA fragment (CDNA5 ISA): VGPR v of a lane holds the
// contiguous K-pair (2v,2v+1) at K = 2v + (v>=4?8:0) + (lane>=16?8:0).
__device__ __forceinline__ int wmma_kpair(int v, int lane) {
  return 2 * v + ((v >= 4) ? 8 : 0) + ((lane >= 16) ? 8 : 0);
}

union BFU { v16h v; unsigned int u[8]; };

__device__ __forceinline__ float sigf(float x) { return 1.0f / (1.0f + __expf(-x)); }

// ---------------------------------------------------------------- utilities
__global__ void k_zero(float* p, int n) {
  int i = blockIdx.x * blockDim.x + threadIdx.x;
  if (i < n) p[i] = 0.0f;
}

// U_I f32 -> f16 with padded stride KP (zero pad); one row per block
__global__ void k_cvt_h(const float* __restrict__ in, _Float16* __restrict__ out) {
  int r = blockIdx.x;
  for (int k = threadIdx.x; k < KP; k += 256)
    out[(size_t)r * KP + k] = (k < N_UI) ? (_Float16)in[(size_t)r * N_UI + k] : (_Float16)0.0f;
}

// out[row] = row * sigmoid(row @ W + b)   (128 threads / row)
__global__ void k_gate(const float* __restrict__ x, const float* __restrict__ W,
                       const float* __restrict__ b, float* __restrict__ out) {
  __shared__ float xs[D];
  int row = blockIdx.x, t = threadIdx.x;
  xs[t] = x[(size_t)row * D + t];
  __syncthreads();
  float s = b[t];
#pragma unroll 4
  for (int k = 0; k < D; ++k) s += xs[k] * W[k * D + t];
  out[(size_t)row * D + t] = xs[t] * sigf(s);
}

// he[r][d] = sum_p PU[r][p] * e[p][d]
__global__ void k_pu(const float* __restrict__ PU, const float* __restrict__ e,
                     float* __restrict__ he) {
  int r = blockIdx.x, d = threadIdx.x;
  float s = 0.0f;
  for (int p = 0; p < POI; ++p) s = fmaf(PU[(size_t)r * POI + p], e[(size_t)p * D + d], s);
  he[r * D + d] = s;
}

// e_out[p][d] = sum_h UP[p][h]*he[h][d];  acc[p][d] += e_out
__global__ void k_up(const float* __restrict__ UP, const float* __restrict__ he,
                     float* __restrict__ e_out, float* __restrict__ acc) {
  __shared__ float hs[N_HE * D];
  for (int i = threadIdx.x; i < N_HE * D; i += blockDim.x) hs[i] = he[i];
  __syncthreads();
  int p = blockIdx.x, d = threadIdx.x;
  float s = 0.0f;
#pragma unroll
  for (int h = 0; h < N_HE; ++h) s = fmaf(UP[(size_t)p * N_HE + h], hs[h * D + d], s);
  e_out[(size_t)p * D + d] = s;
  acc[(size_t)p * D + d] += s;
}

__global__ void k_l2norm(const float* __restrict__ in, float* __restrict__ out) {
  __shared__ float red[D];
  size_t base = (size_t)blockIdx.x * D;
  float v = in[base + threadIdx.x];
  red[threadIdx.x] = v * v; __syncthreads();
  for (int s = 64; s > 0; s >>= 1) { if (threadIdx.x < s) red[threadIdx.x] += red[threadIdx.x + s]; __syncthreads(); }
  float nrm = fmaxf(sqrtf(red[0]), 1e-12f);
  out[base + threadIdx.x] = v / nrm;
}

__global__ void k_add_l2norm(const float* __restrict__ a, const float* __restrict__ b,
                             float* __restrict__ out) {
  __shared__ float red[D];
  size_t base = (size_t)blockIdx.x * D;
  float v = a[base + threadIdx.x] + b[base + threadIdx.x];
  red[threadIdx.x] = v * v; __syncthreads();
  for (int s = 64; s > 0; s >>= 1) { if (threadIdx.x < s) red[threadIdx.x] += red[threadIdx.x + s]; __syncthreads(); }
  float nrm = fmaxf(sqrtf(red[0]), 1e-12f);
  out[base + threadIdx.x] = v / nrm;
}

__global__ void k_axpy(float* __restrict__ acc, const float* __restrict__ x, int n) {
  int i = blockIdx.x * blockDim.x + threadIdx.x;
  if (i < n) acc[i] += x[i];
}

__global__ void k_combine(const float* __restrict__ a, const float* __restrict__ b,
                          float* __restrict__ o, float s, int n) {
  int i = blockIdx.x * blockDim.x + threadIdx.x;
  if (i < n) o[i] = (a[i] + b[i]) * s;
}

// ---- shared staging helper: B chunk (32 x 64) -> Bs[n][k] f16, fast path clause
__device__ __forceinline__ void stage_B(const float* __restrict__ B, int K, int k0,
                                        int colbase, int t, _Float16 (*Bs)[40]) {
  if (k0 + 32 <= K) {
    float2 v2[4];
#pragma unroll
    for (int j = 0; j < 4; ++j) {
      int i2 = (t + j * 256) * 2;
      int nn = i2 & 63, kk = i2 >> 6;
      v2[j] = *(const float2*)&B[(size_t)(k0 + kk) * D + colbase + nn];
    }
#pragma unroll
    for (int j = 0; j < 4; ++j) {
      int i2 = (t + j * 256) * 2;
      int nn = i2 & 63, kk = i2 >> 6;
      Bs[nn][kk]     = (_Float16)v2[j].x;
      Bs[nn + 1][kk] = (_Float16)v2[j].y;
    }
  } else {
    for (int i = t; i < 32 * 64; i += 256) {
      int nn = i & 63, kk = i >> 6;
      float v = (k0 + kk < K) ? B[(size_t)(k0 + kk) * D + colbase + nn] : 0.0f;
      Bs[nn][kk] = (_Float16)v;
    }
  }
}

// ------------------- WMMA GEMM (f32 A): C[Mx128] = A[MxK] @ B[Kx128]
// Tile 128 rows x 64 cols; 8 waves, 4 accumulators each.
__global__ __launch_bounds__(256) void k_gemm_wmma(
    const float* __restrict__ A, const float* __restrict__ B,
    float* __restrict__ C, int M, int K) {
  __shared__ _Float16 Bs[64][40];
  int t = threadIdx.x, wave = t >> 5, lane = t & 31;
  int rowbase = blockIdx.x * 128 + wave * 16;
  int colbase = blockIdx.y * 64;
  int arow = rowbase + (lane & 15);
  int kw4 = (lane >= 16) ? 4 : 0;
  v8f acc[4] = {};
  for (int k0 = 0; k0 < K; k0 += 32) {
    stage_B(B, K, k0, colbase, t, Bs);
    __syncthreads();
    {
      const float* pf = (arow < M && k0 + 32 < K) ? &A[(size_t)arow * K + k0 + 32] : A;
      __builtin_prefetch(pf, 0, 0);
    }
    v16h afrag;
    if (arow < M && k0 + 32 <= K) {          // fast: 8 aligned float2 loads
      float2 p[8];
#pragma unroll
      for (int v = 0; v < 8; ++v)
        p[v] = *(const float2*)&A[(size_t)arow * K + k0 + wmma_kpair(v, lane)];
#pragma unroll
      for (int v = 0; v < 8; ++v) {
        afrag[2 * v]     = (_Float16)p[v].x;
        afrag[2 * v + 1] = (_Float16)p[v].y;
      }
    } else {
#pragma unroll
      for (int v = 0; v < 8; ++v) {
        int kp = k0 + wmma_kpair(v, lane);
        float a0 = (arow < M && kp < K)     ? A[(size_t)arow * K + kp]     : 0.0f;
        float a1 = (arow < M && kp + 1 < K) ? A[(size_t)arow * K + kp + 1] : 0.0f;
        afrag[2 * v]     = (_Float16)a0;
        afrag[2 * v + 1] = (_Float16)a1;
      }
    }
#pragma unroll
    for (int sub = 0; sub < 4; ++sub) {
      BFU bf;
      const unsigned int* brow = (const unsigned int*)&Bs[sub * 16 + (lane & 15)][0];
#pragma unroll
      for (int v = 0; v < 8; ++v) bf.u[v] = brow[v + ((v >= 4) ? 4 : 0) + kw4];
      acc[sub] = __builtin_amdgcn_wmma_f32_16x16x32_f16(
          false, afrag, false, bf.v, (short)0, acc[sub], false, false);
    }
    __syncthreads();
  }
  int nb = lane & 15, mofs = 8 * (lane >> 4);
#pragma unroll
  for (int sub = 0; sub < 4; ++sub)
#pragma unroll
    for (int r = 0; r < 8; ++r) {
      int grow = rowbase + r + mofs;
      if (grow < M) C[(size_t)grow * D + colbase + sub * 16 + nb] = acc[sub][r];
    }
}

// ------------------- WMMA GEMM (f16 A, padded lda): C[Mx128] = A[MxK] @ B[Kx128]
// A fragment = exactly two aligned global b128 loads per chunk.
__global__ __launch_bounds__(256) void k_gemm_wmma_h(
    const _Float16* __restrict__ A, int lda, const float* __restrict__ B,
    float* __restrict__ C, int M, int K) {
  __shared__ _Float16 Bs[64][40];
  int t = threadIdx.x, wave = t >> 5, lane = t & 31;
  int rowbase = blockIdx.x * 128 + wave * 16;
  int colbase = blockIdx.y * 64;
  int arow = rowbase + (lane & 15);
  int kw4 = (lane >= 16) ? 4 : 0;
  v8f acc[4] = {};
  for (int k0 = 0; k0 < K; k0 += 32) {
    stage_B(B, K, k0, colbase, t, Bs);
    __syncthreads();
    const _Float16* arp = &A[(size_t)arow * lda + k0];
    {
      const _Float16* pf = (arow < M && k0 + 32 < K) ? arp + 32 : A;
      __builtin_prefetch(pf, 0, 0);
    }
    v16h afrag;
    if (arow < M && k0 + 32 <= K) {          // fast: 2 aligned b128 loads
      v8h lo = *(const v8h*)(arp + 2 * kw4);
      v8h hi = *(const v8h*)(arp + 16 + 2 * kw4);
#pragma unroll
      for (int j = 0; j < 8; ++j) { afrag[j] = lo[j]; afrag[8 + j] = hi[j]; }
    } else {
#pragma unroll
      for (int v = 0; v < 8; ++v) {
        int kp = k0 + wmma_kpair(v, lane);
        afrag[2 * v]     = (arow < M && kp < K)     ? A[(size_t)arow * lda + kp]     : (_Float16)0.0f;
        afrag[2 * v + 1] = (arow < M && kp + 1 < K) ? A[(size_t)arow * lda + kp + 1] : (_Float16)0.0f;
      }
    }
#pragma unroll
    for (int sub = 0; sub < 4; ++sub) {
      BFU bf;
      const unsigned int* brow = (const unsigned int*)&Bs[sub * 16 + (lane & 15)][0];
#pragma unroll
      for (int v = 0; v < 8; ++v) bf.u[v] = brow[v + ((v >= 4) ? 4 : 0) + kw4];
      acc[sub] = __builtin_amdgcn_wmma_f32_16x16x32_f16(
          false, afrag, false, bf.v, (short)0, acc[sub], false, false);
    }
    __syncthreads();
  }
  int nb = lane & 15, mofs = 8 * (lane >> 4);
#pragma unroll
  for (int sub = 0; sub < 4; ++sub)
#pragma unroll
    for (int r = 0; r < 8; ++r) {
      int grow = rowbase + r + mofs;
      if (grow < M) C[(size_t)grow * D + colbase + sub * 16 + nb] = acc[sub][r];
    }
}

// ---------------------- InfoNCE: streaming sim tile (z1@z2^T)/temp, fixed max=1/temp=5
__global__ __launch_bounds__(256) void k_simnce(
    const float* __restrict__ z1, const float* __restrict__ z2,
    float* __restrict__ rowsum, float* __restrict__ colsum,
    float* __restrict__ diag, int N) {
  __shared__ float rs[128], cs[128];
  int t = threadIdx.x, wave = t >> 5, lane = t & 31;
  if (t < 128) { rs[t] = 0.0f; cs[t] = 0.0f; }
  __syncthreads();
  int rowbase = blockIdx.x * 128 + wave * 16;
  int colbase = blockIdx.y * 128;
  int mrow = rowbase + (lane & 15);
  v16h afrag[4];
  if (mrow < N) {                 // per-fragment guard (ragged only on last tile)
#pragma unroll
    for (int kc = 0; kc < 4; ++kc) {
      float2 p[8];
#pragma unroll
      for (int v = 0; v < 8; ++v)
        p[v] = *(const float2*)&z1[(size_t)mrow * D + kc * 32 + wmma_kpair(v, lane)];
#pragma unroll
      for (int v = 0; v < 8; ++v) {
        afrag[kc][2 * v]     = (_Float16)p[v].x;
        afrag[kc][2 * v + 1] = (_Float16)p[v].y;
      }
    }
  } else {
#pragma unroll
    for (int kc = 0; kc < 4; ++kc)
#pragma unroll
      for (int j = 0; j < 16; ++j) afrag[kc][j] = (_Float16)0.0f;
  }
#pragma unroll
  for (int sub = 0; sub < 8; ++sub) {
    int ncol = colbase + sub * 16 + (lane & 15);
    v8f acc = {};
#pragma unroll
    for (int kc = 0; kc < 4; ++kc) {
      v16h bfrag;
      if (ncol < N) {
        float2 p[8];
#pragma unroll
        for (int v = 0; v < 8; ++v)
          p[v] = *(const float2*)&z2[(size_t)ncol * D + kc * 32 + wmma_kpair(v, lane)];
#pragma unroll
        for (int v = 0; v < 8; ++v) {
          bfrag[2 * v]     = (_Float16)p[v].x;
          bfrag[2 * v + 1] = (_Float16)p[v].y;
        }
      } else {
#pragma unroll
        for (int j = 0; j < 16; ++j) bfrag[j] = (_Float16)0.0f;
      }
      acc = __builtin_amdgcn_wmma_f32_16x16x32_f16(
          false, afrag[kc], false, bfrag, (short)0, acc, false, false);
    }
    int mofs = 8 * (lane >> 4);
#pragma unroll
    for (int r = 0; r < 8; ++r) {
      int grow = rowbase + r + mofs;
      if (grow < N && ncol < N) {
        float v = acc[r] * 5.0f;          // /temp
        float p = __expf(v - 5.0f);       // stable: sim <= 5 (unit vectors)
        atomicAdd(&rs[grow - blockIdx.x * 128], p);
        atomicAdd(&cs[ncol - colbase], p);
        if (grow == ncol) diag[grow] = v;
      }
    }
  }
  __syncthreads();
  if (t < 128) {
    int gr = blockIdx.x * 128 + t, gc = colbase + t;
    if (gr < N) atomicAdd(&rowsum[gr], rs[t]);
    if (gc < N) atomicAdd(&colsum[gc], cs[t]);
  }
}

__global__ void k_nce_finish(const float* __restrict__ rowsum, const float* __restrict__ colsum,
                             const float* __restrict__ diag, int N, float* __restrict__ loss) {
  __shared__ float red[256];
  float s = 0.0f;
  for (int i = threadIdx.x; i < N; i += 256) {
    float lr = 5.0f + __logf(rowsum[i]);
    float lc = 5.0f + __logf(colsum[i]);
    s += 0.5f * (lr + lc) - diag[i];
  }
  red[threadIdx.x] = s; __syncthreads();
  for (int st = 128; st > 0; st >>= 1) { if (threadIdx.x < st) red[threadIdx.x] += red[threadIdx.x + st]; __syncthreads(); }
  if (threadIdx.x == 0) atomicAdd(loss, red[0] / (float)N);
}

// small InfoNCE (42x42), scalar path
__global__ void k_nce_small(const float* __restrict__ z1, const float* __restrict__ z2,
                            float* __restrict__ loss, int n) {
  __shared__ float rs[64], cs[64], dg[64], red[256];
  int t = threadIdx.x;
  if (t < 64) { rs[t] = 0.0f; cs[t] = 0.0f; dg[t] = 0.0f; }
  __syncthreads();
  for (int idx = t; idx < n * n; idx += 256) {
    int i = idx / n, j = idx % n;
    float s = 0.0f;
#pragma unroll 8
    for (int d = 0; d < D; ++d) s = fmaf(z1[i * D + d], z2[j * D + d], s);
    s *= 5.0f;
    float p = __expf(s - 5.0f);
    atomicAdd(&rs[i], p); atomicAdd(&cs[j], p);
    if (i == j) dg[i] = s;
  }
  __syncthreads();
  float a = 0.0f;
  for (int i = t; i < n; i += 256)
    a += 0.5f * ((5.0f + __logf(rs[i])) + (5.0f + __logf(cs[i]))) - dg[i];
  red[t] = a; __syncthreads();
  for (int st = 128; st > 0; st >>= 1) { if (t < st) red[t] += red[t + st]; __syncthreads(); }
  if (t == 0) atomicAdd(loss, red[0] / (float)n);
}

// ---------------------------------------------------------------- gated KAN pieces
__global__ void k_twiddle(float* __restrict__ Ct, float* __restrict__ St) {
  int n = blockIdx.x, k = threadIdx.x;
  float ang = 6.283185307179586f * (float)(n * k) / 128.0f;
  Ct[n * D + k] = __cosf(ang);
  St[n * D + k] = -__sinf(ang);
}

__global__ void k_cheb_T(const float* __restrict__ xa, const float* __restrict__ xb,
                         float* __restrict__ T, int M) {
  int idx = blockIdx.x * blockDim.x + threadIdx.x;
  if (idx >= M * D) return;
  int n = idx / D, i = idx % D;
  float a = tanhf(xa[idx]), b = tanhf(xb[idx]);
  float a2 = 2.0f * a * a - 1.0f, b2 = 2.0f * b * b - 1.0f;
  float* tp = &T[(size_t)n * (D * 4) + i * 4];
  tp[0] = 2.0f;
  tp[1] = a + b;
  tp[2] = a2 + b2;
  tp[3] = (2.0f * a * a2 - a) + (2.0f * b * b2 - b);
}

__global__ void k_packW(const float* __restrict__ cheb, float* __restrict__ W) {
  int i = blockIdx.x, o = threadIdx.x;
#pragma unroll
  for (int k = 0; k < 4; ++k)
    W[(size_t)(i * 4 + k) * D + o] = cheb[((size_t)i * D + o) * 4 + k];
}

__global__ void k_sig_mix(const float* __restrict__ gl, const float* __restrict__ x1,
                          const float* __restrict__ x2, float* __restrict__ out, int n) {
  int i = blockIdx.x * blockDim.x + threadIdx.x;
  if (i >= n) return;
  float g = sigf(gl[i]);
  out[i] = g * x1[i] + (1.0f - g) * x2[i];
}

// ---------------------------------------------------------------- tail
__global__ void k_final_u(const float* __restrict__ ue, const float* __restrict__ hgu,
                          const int* __restrict__ user_ui, const int* __restrict__ user_col,
                          float* __restrict__ fu) {
  int d = threadIdx.x;
  fu[d] = ue[(size_t)(*user_ui) * D + d] + hgu[(size_t)(*user_col) * D + d];
}

__global__ void k_matvec(const float* __restrict__ P, const float* __restrict__ fu,
                         float* __restrict__ out) {
  __shared__ float red[D];
  int row = blockIdx.x;
  red[threadIdx.x] = P[(size_t)row * D + threadIdx.x] * fu[threadIdx.x];
  __syncthreads();
  for (int s = 64; s > 0; s >>= 1) { if (threadIdx.x < s) red[threadIdx.x] += red[threadIdx.x + s]; __syncthreads(); }
  if (threadIdx.x == 0) out[row] = red[0];
}

__global__ void k_pred_loss(const float* __restrict__ logits, const int* __restrict__ target,
                            float* __restrict__ loss, int n) {
  __shared__ float red[256];
  float m = -1e30f;
  for (int i = threadIdx.x; i < n; i += 256) m = fmaxf(m, logits[i]);
  red[threadIdx.x] = m; __syncthreads();
  for (int s = 128; s > 0; s >>= 1) { if (threadIdx.x < s) red[threadIdx.x] = fmaxf(red[threadIdx.x], red[threadIdx.x + s]); __syncthreads(); }
  float gmax = red[0]; __syncthreads();
  float sum = 0.0f;
  for (int i = threadIdx.x; i < n; i += 256) sum += __expf(logits[i] - gmax);
  red[threadIdx.x] = sum; __syncthreads();
  for (int s = 128; s > 0; s >>= 1) { if (threadIdx.x < s) red[threadIdx.x] += red[threadIdx.x + s]; __syncthreads(); }
  if (threadIdx.x == 0) atomicAdd(loss, __logf(red[0]) + gmax - logits[*target]);
}

__global__ void k_store_loss(const float* __restrict__ scal, float* __restrict__ out) {
  out[0] = scal[0];
}

// ==========================================================================
extern "C" void kernel_launch(void* const* d_in, const int* in_sizes, int n_in,
                              void* d_out, int out_size, void* d_ws, size_t ws_size,
                              hipStream_t stream) {
  (void)in_sizes; (void)n_in; (void)out_size; (void)ws_size;
  const float* poi_w  = (const float*)d_in[0];
  const float* ui_w   = (const float*)d_in[1];
  const float* w_gc   = (const float*)d_in[2];
  const float* b_gc   = (const float*)d_in[3];
  const float* w_gu   = (const float*)d_in[4];
  const float* b_gu   = (const float*)d_in[5];
  const float* cheb_r = (const float*)d_in[6];
  const float* cheb_i = (const float*)d_in[7];
  const float* HG_pu  = (const float*)d_in[8];
  const float* HG_up  = (const float*)d_in[9];
  const float* U_I    = (const float*)d_in[10];
  const float* noise1 = (const float*)d_in[11];
  const float* noise2 = (const float*)d_in[12];
  const int*   target   = (const int*)d_in[13];
  const int*   user_col = (const int*)d_in[14];
  const int*   user_ui  = (const int*)d_in[15];
  float* out = (float*)d_out;   // [0]=loss, [1..POI]=output logits
  float* logits = out + 1;

  char* cur = (char*)d_ws;
  auto alloc = [&](size_t n) { float* p = (float*)cur; cur += n * sizeof(float); return p; };
  float* cg   = alloc((size_t)POI * D);
  float* ug   = alloc((size_t)N_UI * D);
  float* hacc = alloc((size_t)POI * D);
  float* ebuf = alloc((size_t)POI * D);
  float* he   = alloc((size_t)N_HE * D);
  float* hg   = alloc((size_t)POI * D);
  float* z2   = alloc((size_t)POI * D);
  float* hgu  = alloc((size_t)N_HE * D);
  float* uacc = alloc((size_t)N_UI * D);
  float* ue1  = alloc((size_t)N_UI * D);
  float* ue2  = alloc((size_t)N_UI * D);
  float* ue   = alloc((size_t)N_UI * D);
  float* z2u  = alloc((size_t)N_UI * D);
  float* rs_  = alloc(N_UI);
  float* cs_  = alloc(N_UI);
  float* dg_  = alloc(N_UI);
  float* Ct   = alloc((size_t)D * D);
  float* St   = alloc((size_t)D * D);
  float* X1r  = alloc((size_t)POI * D);
  float* X1i  = alloc((size_t)POI * D);
  float* X2r  = alloc((size_t)POI * D);
  float* X2i  = alloc((size_t)POI * D);
  float* Tbuf = alloc((size_t)POI * D * 4);
  float* Wbig = alloc((size_t)D * 4 * D);
  float* Gl   = alloc((size_t)POI * D);
  float* Fr   = alloc((size_t)POI * D);
  float* Fi   = alloc((size_t)POI * D);
  float* Pa   = alloc((size_t)POI * D);
  float* Pb   = alloc((size_t)POI * D);
  float* Fp   = alloc((size_t)POI * D);
  float* znc  = alloc((size_t)N_UC * D);
  float* znl  = alloc((size_t)N_UC * D);
  float* fu   = alloc(D);
  float* scal = alloc(8);
  cur = (char*)(((uintptr_t)cur + 255) & ~(uintptr_t)255);
  _Float16* Uh = (_Float16*)cur; cur += (size_t)N_UI * KP * sizeof(_Float16);

  auto gemm = [&](const float* A, const float* B, float* C, int M, int K) {
    k_gemm_wmma<<<dim3((M + 127) / 128, 2), 256, 0, stream>>>(A, B, C, M, K);
  };
  auto gemm_h = [&](const _Float16* A, int lda, const float* B, float* C, int M, int K) {
    k_gemm_wmma_h<<<dim3((M + 127) / 128, 2), 256, 0, stream>>>(A, lda, B, C, M, K);
  };
  auto nce = [&](const float* za, const float* zb, int N) {
    int nb = (N + 255) / 256;
    k_zero<<<nb, 256, 0, stream>>>(rs_, N);
    k_zero<<<nb, 256, 0, stream>>>(cs_, N);
    k_zero<<<nb, 256, 0, stream>>>(dg_, N);
    int tiles = (N + 127) / 128;
    k_simnce<<<dim3(tiles, tiles), 256, 0, stream>>>(za, zb, rs_, cs_, dg_, N);
    k_nce_finish<<<1, 256, 0, stream>>>(rs_, cs_, dg_, N, scal);
  };

  k_zero<<<1, 256, 0, stream>>>(scal, 8);

  // --- gated embeddings
  k_gate<<<POI, D, 0, stream>>>(poi_w, w_gc, b_gc, cg);       // col_gate
  k_gate<<<N_UI, D, 0, stream>>>(ui_w, w_gu, b_gu, ug);       // U_gate

  // --- convert U_I to f16 (135.7 MB: fits in 192 MB L2 for the 2nd layer)
  k_cvt_h<<<N_UI, 256, 0, stream>>>(U_I, Uh);

  // --- hconv (2 layers) + l2norm  (1/(L+1) scale dropped: invariant under l2norm)
  hipMemcpyAsync(hacc, cg, (size_t)POI * D * sizeof(float), hipMemcpyDeviceToDevice, stream);
  k_pu<<<N_HE, D, 0, stream>>>(HG_pu, cg, he);
  k_up<<<POI, D, 0, stream>>>(HG_up, he, ebuf, hacc);
  k_pu<<<N_HE, D, 0, stream>>>(HG_pu, ebuf, he);
  k_up<<<POI, D, 0, stream>>>(HG_up, he, ebuf, hacc);
  k_l2norm<<<POI, D, 0, stream>>>(hacc, hg);

  // --- level_loss = InfoNCE(hg, l2norm(hg + noise1))
  k_add_l2norm<<<POI, D, 0, stream>>>(hg, noise1, z2);
  nce(hg, z2, POI);

  // --- hg_users = HG_pu @ hg
  k_pu<<<N_HE, D, 0, stream>>>(HG_pu, hg, hgu);

  // --- gconv (2 layers of U_I @ e, f16-A WMMA) + l2norm
  hipMemcpyAsync(uacc, ug, (size_t)N_UI * D * sizeof(float), hipMemcpyDeviceToDevice, stream);
  gemm_h(Uh, KP, ug, ue1, N_UI, N_UI);
  k_axpy<<<(N_UI * D + 255) / 256, 256, 0, stream>>>(uacc, ue1, N_UI * D);
  gemm_h(Uh, KP, ue1, ue2, N_UI, N_UI);
  k_axpy<<<(N_UI * D + 255) / 256, 256, 0, stream>>>(uacc, ue2, N_UI * D);
  k_l2norm<<<N_UI, D, 0, stream>>>(uacc, ue);

  // --- level_loss1 = InfoNCE(ue, l2norm(ue + noise2))
  k_add_l2norm<<<N_UI, D, 0, stream>>>(ue, noise2, z2u);
  nce(ue, z2u, N_UI);

  const float* poi_e = ue + (size_t)N_UC * D;   // rows already unit-norm

  // --- ssl_p = InfoNCE(hg, poi_e)
  nce(hg, poi_e, POI);

  // --- ssl = InfoNCE(user_c = ue[:42], user_cl = hg_users[9:])
  k_l2norm<<<N_UC, D, 0, stream>>>(ue, znc);
  k_l2norm<<<N_UC, D, 0, stream>>>(hgu + 9 * D, znl);
  k_nce_small<<<1, 256, 0, stream>>>(znc, znl, scal, N_UC);

  // --- gated KAN: FFT via DFT-matrix GEMMs
  k_twiddle<<<D, D, 0, stream>>>(Ct, St);
  gemm(hg,    Ct, X1r, POI, D);
  gemm(hg,    St, X1i, POI, D);
  gemm(poi_e, Ct, X2r, POI, D);
  gemm(poi_e, St, X2i, POI, D);

  int nElem = POI * D;
  // real gate + mix
  k_packW<<<D, D, 0, stream>>>(cheb_r, Wbig);
  k_cheb_T<<<(nElem + 255) / 256, 256, 0, stream>>>(X1r, X2r, Tbuf, POI);
  gemm(Tbuf, Wbig, Gl, POI, D * 4);
  k_sig_mix<<<(nElem + 255) / 256, 256, 0, stream>>>(Gl, X1r, X2r, Fr, nElem);
  // imag gate + mix
  k_packW<<<D, D, 0, stream>>>(cheb_i, Wbig);
  k_cheb_T<<<(nElem + 255) / 256, 256, 0, stream>>>(X1i, X2i, Tbuf, POI);
  gemm(Tbuf, Wbig, Gl, POI, D * 4);
  k_sig_mix<<<(nElem + 255) / 256, 256, 0, stream>>>(Gl, X1i, X2i, Fi, nElem);
  // ifft real part: (Fr@cos + Fi@(-sin)) / 128
  gemm(Fr, Ct, Pa, POI, D);
  gemm(Fi, St, Pb, POI, D);
  k_combine<<<(nElem + 255) / 256, 256, 0, stream>>>(Pa, Pb, Fp, 1.0f / 128.0f, nElem);

  // --- output = final_p @ (ue[user_ui] + hg_users[user_col]); loss tail
  k_final_u<<<1, D, 0, stream>>>(ue, hgu, user_ui, user_col, fu);
  k_matvec<<<POI, D, 0, stream>>>(Fp, fu, logits);
  k_pred_loss<<<1, 256, 0, stream>>>(logits, target, scal, POI);
  k_store_loss<<<1, 1, 0, stream>>>(scal, out);
}
